// theta_solver_35931696398552
// MI455X (gfx1250) — compile-verified, matrix operand order checked by reading, not run on hardware
//
#include <hip/hip_runtime.h>
#include <math.h>

// ---------------- problem constants ----------------
#define B_ROWS  65536
#define D_DIM   256
#define HID_DIM 512
#define NSTEPS  8
#define FPITERS 20
#define HSTEP   0.125f   // T_FINAL / NUM_STEPS
#define THETA   0.5f

// ---------------- tiling ----------------
#define BM      64            // rows per workgroup tile
#define THREADS 256           // 8 waves (wave32)
// LDS row pitches (elements); pitch*elemsize % 16 == 0, staggered vs banks
#define PY 264                // yb  (bf16)  pitch: 528 B
#define PH 520                // hb  (bf16)  pitch: 1040 B
#define PE 260                // f32 buffers pitch: 1040 B

typedef __attribute__((ext_vector_type(16))) __bf16 v16bf;
typedef __attribute__((ext_vector_type(8)))  __bf16 v8bf;
typedef __attribute__((ext_vector_type(8)))  float  v8f;

__device__ __forceinline__ v16bf cat8(v8bf lo, v8bf hi) {
    return __builtin_shufflevector(lo, hi, 0,1,2,3,4,5,6,7,8,9,10,11,12,13,14,15);
}

// Native v_tanh_f32 on gfx1250 (probe-confirmed last round); branch-free fallback.
__device__ __forceinline__ float fast_tanh(float x) {
#if __has_builtin(__builtin_amdgcn_tanhf)
    return __builtin_amdgcn_tanhf(x);
#else
    float ax = __builtin_fabsf(x);
    float e  = __builtin_amdgcn_exp2f(ax * -2.8853900817779268f);
    float r  = (1.0f - e) * __builtin_amdgcn_rcpf(1.0f + e);
    return __builtin_copysignf(r, x);
#endif
}

// ---------------- weight prep: transpose + fp32 -> bf16 ----------------
// W1 [D][HID] -> W1T [HID][D];  W2 [HID][D] -> W2T [D][HID]
__global__ void prep_weights(const float* __restrict__ W1,
                             const float* __restrict__ W2,
                             __bf16* __restrict__ W1T,
                             __bf16* __restrict__ W2T) {
    int i = blockIdx.x * blockDim.x + threadIdx.x;
    if (i < D_DIM * HID_DIM) {
        int h = i / D_DIM, d = i % D_DIM;           // W1T[h][d] = W1[d][h]
        W1T[i] = (__bf16)W1[d * HID_DIM + h];
        int d2 = i / HID_DIM, h2 = i % HID_DIM;     // W2T[d][h] = W2[h][d]
        W2T[i] = (__bf16)W2[h2 * D_DIM + d2];
    }
}

// ---------------- main persistent theta-method kernel ----------------
// Each wave covers ALL 64 rows (4 x 16-row strips) of the block tile for a
// disjoint N-slice: every weight tile is loaded exactly once per block per
// f-eval (minimum L2 traffic). GEMM1 is done in two N-passes to keep the
// accumulator live-set at 64 VGPRs (total pressure < 256, no VGPR-MSB toggles).
__global__ __launch_bounds__(THREADS, 1)
void theta_solver_kernel(const float* __restrict__ x,
                         const __bf16* __restrict__ W1T,   // [HID][D]
                         const __bf16* __restrict__ W2T,   // [D][HID]
                         const float* __restrict__ b1,
                         const float* __restrict__ b2,
                         float* __restrict__ out) {
    __shared__ __align__(16) __bf16 yb[BM * PY];      // bf16 current iterate z
    __shared__ __align__(16) __bf16 hb[BM * PH];      // bf16 tanh hidden act
    __shared__ __align__(16) float  expl_s[BM * PE];  // y + h(1-theta) f(y)
    __shared__ __align__(16) float  ycur[BM * PE];    // fp32 state y
    __shared__ __align__(16) float  b1s[HID_DIM];
    __shared__ __align__(16) float  b2s[D_DIM];

    const int tid   = threadIdx.x;
    const int lane  = tid & 31;
    const int wave  = tid >> 5;     // 0..7, owns an N-slice
    const int l15   = lane & 15;
    const int hsel  = lane >> 4;    // lane half (K +8 selector)
    const int row0  = blockIdx.x * BM;
    const int nb1   = wave * 64;    // GEMM1 N-slice base (4 tiles of 16)
    const int nb2   = wave * 32;    // GEMM2 N-slice base (2 tiles of 16)

    // ---- load biases + input tile ----
    for (int i = tid; i < HID_DIM; i += THREADS) b1s[i] = b1[i];
    for (int i = tid; i < D_DIM;  i += THREADS) b2s[i] = b2[i];
    for (int i = tid; i < BM * D_DIM; i += THREADS) {
        int r = i >> 8, c = i & 255;
        float v = x[(size_t)(row0 + r) * D_DIM + c];
        ycur[r * PE + c] = v;
        yb[r * PY + c]   = (__bf16)v;
    }
    __syncthreads();

#pragma clang loop unroll(disable)
    for (int step = 0; step < NSTEPS; ++step) {
#pragma clang loop unroll(disable)
        for (int it = 0; it < FPITERS; ++it) {
            // ============ GEMM1: hb = tanh(yb @ W1 + b1) ============
            // M = 64 (4 strips), N-slice = 64, K = D = 256 (8 steps)
            // Two passes of 2 n-tiles -> 64 accumulator VGPRs live.
#pragma clang loop unroll(disable)
            for (int p = 0; p < 2; ++p) {
                v8f acc1[2][4] = {};                  // [n-tile][m-strip]
#pragma clang loop unroll(disable)
                for (int k = 0; k < 8; ++k) {
                    v16bf a[4];
#pragma unroll
                    for (int m = 0; m < 4; ++m) {
                        const __bf16* pa = &yb[(m * 16 + l15) * PY + k * 32 + hsel * 8];
                        a[m] = cat8(*(const v8bf*)pa, *(const v8bf*)(pa + 16));
                    }
#pragma unroll
                    for (int t = 0; t < 2; ++t) {
                        const __bf16* wp = W1T +
                            (size_t)(nb1 + (p * 2 + t) * 16 + l15) * D_DIM + k * 32 + hsel * 8;
                        v16bf bm = cat8(*(const v8bf*)wp, *(const v8bf*)(wp + 16));
#pragma unroll
                        for (int m = 0; m < 4; ++m)
                            acc1[t][m] = __builtin_amdgcn_wmma_f32_16x16x32_bf16(
                                false, a[m], false, bm, (short)0, acc1[t][m], false, false);
                    }
                }
                // epilogue: bias + native tanh -> hb (bf16)
#pragma unroll
                for (int t = 0; t < 2; ++t) {
                    const int n = nb1 + (p * 2 + t) * 16 + l15;
                    const float bias = b1s[n];
#pragma unroll
                    for (int m = 0; m < 4; ++m) {
                        const int rb = m * 16 + hsel * 8;
#pragma unroll
                        for (int v = 0; v < 8; ++v)
                            hb[(rb + v) * PH + n] =
                                (__bf16)fast_tanh(acc1[t][m][v] + bias);
                    }
                }
            }
            __syncthreads();

            // ======== GEMM2 + theta update: F = hb @ W2 + b2 ========
            // M = 64 (4 strips), N-slice = 32 (2 tiles), K = HID = 512 (16 steps)
            v8f acc2[2][4] = {};
#pragma clang loop unroll(disable)
            for (int k = 0; k < 16; ++k) {
                v16bf a[4];
#pragma unroll
                for (int m = 0; m < 4; ++m) {
                    const __bf16* pa = &hb[(m * 16 + l15) * PH + k * 32 + hsel * 8];
                    a[m] = cat8(*(const v8bf*)pa, *(const v8bf*)(pa + 16));
                }
#pragma unroll
                for (int t = 0; t < 2; ++t) {
                    const __bf16* wp =
                        W2T + (size_t)(nb2 + t * 16 + l15) * HID_DIM + k * 32 + hsel * 8;
                    v16bf bm = cat8(*(const v8bf*)wp, *(const v8bf*)(wp + 16));
#pragma unroll
                    for (int m = 0; m < 4; ++m)
                        acc2[t][m] = __builtin_amdgcn_wmma_f32_16x16x32_bf16(
                            false, a[m], false, bm, (short)0, acc2[t][m], false, false);
                }
            }
            // epilogue: theta-method update
            const bool last = (it == FPITERS - 1);
#pragma unroll
            for (int t = 0; t < 2; ++t) {
                const int n = nb2 + t * 16 + l15;
                const float bias = b2s[n];
#pragma unroll
                for (int m = 0; m < 4; ++m) {
                    const int rb = m * 16 + hsel * 8;
#pragma unroll
                    for (int v = 0; v < 8; ++v) {
                        const int r = rb + v;
                        const float F = acc2[t][m][v] + bias;
                        float z;
                        if (it == 0) {
                            // fused: f(z0)=f(y). expl = y + h(1-theta)F ; z1 = y + hF
                            const float y = ycur[r * PE + n];
                            expl_s[r * PE + n] = y + HSTEP * (1.0f - THETA) * F;
                            z = y + HSTEP * F;
                        } else {
                            z = expl_s[r * PE + n] + HSTEP * THETA * F;
                        }
                        yb[r * PY + n] = (__bf16)z;
                        if (last) ycur[r * PE + n] = z;   // commit y_{n+1}
                    }
                }
            }
            __syncthreads();
        }
    }

    // ---- write result tile ----
    for (int i = tid; i < BM * D_DIM; i += THREADS) {
        int r = i >> 8, c = i & 255;
        out[(size_t)(row0 + r) * D_DIM + c] = ycur[r * PE + c];
    }
}

// ---------------- launch ----------------
extern "C" void kernel_launch(void* const* d_in, const int* in_sizes, int n_in,
                              void* d_out, int out_size, void* d_ws, size_t ws_size,
                              hipStream_t stream) {
    const float* x  = (const float*)d_in[0];
    const float* W1 = (const float*)d_in[1];
    const float* b1 = (const float*)d_in[2];
    const float* W2 = (const float*)d_in[3];
    const float* b2 = (const float*)d_in[4];

    __bf16* W1T = (__bf16*)d_ws;                       // 512*256 bf16 = 256 KB
    __bf16* W2T = W1T + (size_t)D_DIM * HID_DIM;       // 256*512 bf16 = 256 KB

    const int nw = D_DIM * HID_DIM;
    prep_weights<<<(nw + 255) / 256, 256, 0, stream>>>(W1, W2, W1T, W2T);

    theta_solver_kernel<<<B_ROWS / BM, THREADS, 0, stream>>>(
        x, W1T, W2T, b1, b2, (float*)d_out);
}
